// ModulationConv2D_39333310497100
// MI455X (gfx1250) — compile-verified, matrix operand order checked by reading, not run on hardware
//
#include <hip/hip_runtime.h>
#include <hip/hip_bf16.h>

typedef __attribute__((ext_vector_type(16))) _Float16 v16h;
typedef __attribute__((ext_vector_type(8)))  _Float16 v8h;
typedef __attribute__((ext_vector_type(2)))  _Float16 v2h;
typedef __attribute__((ext_vector_type(8)))  float    v8f;

#define Bn   16
#define Hn   64
#define Wn   64
#define CINc 256
#define COUTc 256
#define SFc  512

// c_dense = sqrt(2/512) = 1/16 exactly; c_conv = sqrt(2/(64*64*256)) = sqrt(2)/1024
#define C_DENSE 0.0625f
#define C_CONV  0.0013810679320049757f

// ---------------------------------------------------------------------------
// 1) s[b,cin] = style[b,:] @ (dense_w * c_dense) + dense_b      (tiny GEMM)
// ---------------------------------------------------------------------------
__global__ void style_dense_kernel(const float* __restrict__ style,
                                   const float* __restrict__ dense_w,
                                   const float* __restrict__ dense_b,
                                   float* __restrict__ s_out) {
    int idx = blockIdx.x * blockDim.x + threadIdx.x;   // B*CIN = 4096
    int b = idx >> 8, c = idx & 255;
    const float* st = style + b * SFc;
    float acc = 0.f;
    for (int f = 0; f < SFc; ++f)
        acc = fmaf(st[f], dense_w[f * CINc + c], acc);
    s_out[idx] = acc * C_DENSE + dense_b[c];
}

// ---------------------------------------------------------------------------
// 2) Pack kernel f32 [tap][cin][cout] -> f16 [tap][cinblk][cout][cin32]
//    so each B-fragment lane load is 32 contiguous bytes (2x b128).
// ---------------------------------------------------------------------------
__global__ void pack_kernel_f16(const float* __restrict__ kernel,
                                _Float16* __restrict__ kf16) {
    int idx = blockIdx.x * blockDim.x + threadIdx.x;   // 9*256*256
    int t   = idx / (CINc * COUTc);
    int r   = idx % (CINc * COUTc);
    int cin = r / COUTc;
    int co  = r % COUTc;
    int dst = (((t * 8 + (cin >> 5)) * COUTc + co) << 5) + (cin & 31);
    kf16[dst] = (_Float16)kernel[idx];
}

// ---------------------------------------------------------------------------
// 3) scale[b,co] = c_conv * rsqrt(c_conv^2 * sum_cin s^2 * (sum_tap k^2) + eps)
// ---------------------------------------------------------------------------
__global__ void demod_scale_kernel(const float* __restrict__ kernel,
                                   const float* __restrict__ s,
                                   float* __restrict__ scale) {
    int idx = blockIdx.x * blockDim.x + threadIdx.x;   // B*COUT = 4096
    int b = idx >> 8, co = idx & 255;
    float acc = 0.f;
    for (int cin = 0; cin < CINc; ++cin) {
        float k2 = 0.f;
        #pragma unroll
        for (int t = 0; t < 9; ++t) {
            float k = kernel[(t * CINc + cin) * COUTc + co];
            k2 = fmaf(k, k, k2);
        }
        float sv = s[b * CINc + cin];
        acc = fmaf(sv * sv, k2, acc);
    }
    scale[idx] = C_CONV * rsqrtf(C_CONV * C_CONV * acc + 1e-8f);
}

// ---------------------------------------------------------------------------
// 4) Main conv: implicit GEMM, one WG per (b,h) row. M=64 pixels, N=256 cout.
//    8 waves: wave wv owns N slice [wv*32, wv*32+32) -> 4 (M) x 2 (N) WMMA tiles.
//    K loop: kh(3) x cinblk(8) [stage modulated f16 slab to LDS] x kw(3).
// ---------------------------------------------------------------------------
#define SLAB_STRIDE 40   // halves per slab row (80B): spreads LDS banks, keeps 16B align

__launch_bounds__(256)
__global__ void modconv_wmma_kernel(const float* __restrict__ x,
                                    const _Float16* __restrict__ kf16,
                                    const float* __restrict__ s,
                                    const float* __restrict__ scale,
                                    float* __restrict__ out) {
    __shared__ _Float16 Aslab[66 * SLAB_STRIDE];   // halo cols -1..64, 32 ch, f16

    const int b = blockIdx.x >> 6;
    const int h = blockIdx.x & 63;

    const int tid   = threadIdx.x;
    const int lane  = tid & 31;
    const int wv    = tid >> 5;       // 0..7
    const int nbase = wv * 32;
    const int lrow  = lane & 15;      // M-row / N-col within 16x16 subtile
    const int khalf = lane >> 4;      // K half: lanes 0-15 -> K 0..15, 16-31 -> K 16..31

    v8f acc[4][2];
    #pragma unroll
    for (int i = 0; i < 4; ++i)
        #pragma unroll
        for (int j = 0; j < 2; ++j)
            acc[i][j] = (v8f){0.f, 0.f, 0.f, 0.f, 0.f, 0.f, 0.f, 0.f};

    const float* sb = s + b * CINc;

    for (int kh = 0; kh < 3; ++kh) {
        const int hh = h + kh - 1;
        const bool rowok = (hh >= 0) && (hh < Hn);
        const float* xrow = x + ((size_t)(b * Hn + hh) * Wn) * CINc;

        for (int cb = 0; cb < 8; ++cb) {
            __syncthreads();   // previous K-steps done reading Aslab
            // ---- stage modulated input slab: 66 cols x 16 half2 pairs ----
            for (int p = tid; p < 66 * 16; p += 256) {
                int row  = p >> 4;          // slab row = wcol + 1
                int cp   = p & 15;
                int wcol = row - 1;
                float2 v = make_float2(0.f, 0.f);
                if (rowok && (unsigned)wcol < (unsigned)Wn) {
                    const float* px = xrow + (size_t)wcol * CINc + cb * 32 + cp * 2;
                    v = *(const float2*)px;
                    if (cb < 7)  // speculative prefetch of next channel block
                        __builtin_prefetch(px + 32, 0, 1);
                }
                float s0 = sb[cb * 32 + cp * 2];
                float s1 = sb[cb * 32 + cp * 2 + 1];
                v2h hv = { (_Float16)(v.x * s0), (_Float16)(v.y * s1) };
                *(v2h*)&Aslab[row * SLAB_STRIDE + cp * 2] = hv;
            }
            __syncthreads();   // slab visible to all waves

            // ---- 3 taps (kw) against this slab: 3 WMMA K-steps ----
            #pragma unroll
            for (int kw = 0; kw < 3; ++kw) {
                const int tap = kh * 3 + kw;
                // B fragments (streamed from L2-resident packed f16 kernel)
                v16h bf[2];
                #pragma unroll
                for (int j = 0; j < 2; ++j) {
                    const _Float16* bp = kf16 +
                        ((((size_t)(tap * 8 + cb)) * COUTc + (nbase + j * 16 + lrow)) << 5)
                        + khalf * 16;
                    v8h lo = *(const v8h*)bp;
                    v8h hi = *(const v8h*)(bp + 8);
                    #pragma unroll
                    for (int e = 0; e < 8; ++e) { bf[j][e] = lo[e]; bf[j][8 + e] = hi[e]; }
                }
                // A fragments from LDS, 8 WMMAs
                #pragma unroll
                for (int i = 0; i < 4; ++i) {
                    const _Float16* ap =
                        &Aslab[(i * 16 + lrow + kw) * SLAB_STRIDE + khalf * 16];
                    v8h lo = *(const v8h*)ap;
                    v8h hi = *(const v8h*)(ap + 8);
                    v16h af;
                    #pragma unroll
                    for (int e = 0; e < 8; ++e) { af[e] = lo[e]; af[8 + e] = hi[e]; }
                    #pragma unroll
                    for (int j = 0; j < 2; ++j)
                        acc[i][j] = __builtin_amdgcn_wmma_f32_16x16x32_f16(
                            false, af, false, bf[j], (short)0, acc[i][j], false, false);
                }
            }
        }
    }

    // ---- epilogue: scale by c_conv * demod, scatter f32 per C layout ----
    const float* scb  = scale + b * COUTc;
    float*       outb = out + ((size_t)(b * Hn + h) * Wn) * COUTc;
    #pragma unroll
    for (int j = 0; j < 2; ++j) {
        const int co = nbase + j * 16 + lrow;
        const float sc = scb[co];
        #pragma unroll
        for (int i = 0; i < 4; ++i) {
            #pragma unroll
            for (int v = 0; v < 8; ++v) {
                int m = i * 16 + v + khalf * 8;     // pixel w
                outb[(size_t)m * COUTc + co] = acc[i][j][v] * sc;
            }
        }
    }
}

// ---------------------------------------------------------------------------
extern "C" void kernel_launch(void* const* d_in, const int* in_sizes, int n_in,
                              void* d_out, int out_size, void* d_ws, size_t ws_size,
                              hipStream_t stream) {
    const float* x       = (const float*)d_in[0];
    const float* style   = (const float*)d_in[1];
    const float* kernel  = (const float*)d_in[2];
    const float* dense_w = (const float*)d_in[3];
    const float* dense_b = (const float*)d_in[4];
    float* out = (float*)d_out;

    char* ws = (char*)d_ws;
    float*    s_buf     = (float*)ws;                 // 16 KB
    float*    scale_buf = (float*)(ws + 16384);       // 16 KB
    _Float16* kf16      = (_Float16*)(ws + 32768);    // 9*256*256*2 = 1.18 MB

    style_dense_kernel<<<(Bn * CINc) / 256, 256, 0, stream>>>(style, dense_w, dense_b, s_buf);
    pack_kernel_f16<<<(9 * CINc * COUTc) / 256, 256, 0, stream>>>(kernel, kf16);
    demod_scale_kernel<<<(Bn * COUTc) / 256, 256, 0, stream>>>(kernel, s_buf, scale_buf);
    modconv_wmma_kernel<<<Bn * Hn, 256, 0, stream>>>(x, kf16, s_buf, scale_buf, out);
}
